// ReplayEncoderT_7258494730788
// MI455X (gfx1250) — compile-verified, matrix operand order checked by reading, not run on hardware
//
#include <hip/hip_runtime.h>
#include <hip/hip_bf16.h>

// ---------------------------------------------------------------------------
// Transformer encoder (B=4, T=2048, D=512, H=8, FF=2048, L=4) for MI455X.
// bf16-resident operands, layout-matched b128 fragment loads, fp32 WMMA accum.
// GEMM: 32x512 tile/block, A panel double-buffered in LDS via
// global_load_async_to_lds_b128 (ASYNCcnt), B panels streamed as b128 loads.
// ---------------------------------------------------------------------------

#define B_   4
#define T_   2048
#define IN_  64
#define D_   512
#define H_   8
#define DH_  64
#define FF_  2048
#define L_   4
#define LAT_ 128
#define PAST_ 40
#define FUT_  120
#define EPS_  1e-5f
#define M_   (B_ * T_)          // 8192 token rows

typedef __attribute__((ext_vector_type(16))) __bf16 v16bf;
typedef __attribute__((ext_vector_type(8)))  float  v8f;
typedef int v4i_ __attribute__((vector_size(16)));

__device__ __forceinline__ unsigned short f2bfu(float f) {
  unsigned u = __builtin_bit_cast(unsigned, f);
  u += 0x7fffu + ((u >> 16) & 1u);          // round-to-nearest-even
  return (unsigned short)(u >> 16);
}

// CDNA5 16-bit fragment (ISA 05_wmma §7.12.2): lane half hi holds
// K = {8hi..8hi+7} ∪ {16+8hi..16+8hi+7}; fragment == two contiguous 16-byte
// chunks of a K-major bf16 row.
__device__ __forceinline__ v16bf make_frag(const unsigned short* p, int hi) {
  union { v16bf v; uint4 q[2]; } u;
  u.q[0] = *(const uint4*)(p + 8 * hi);
  u.q[1] = *(const uint4*)(p + 16 + 8 * hi);
  return u.v;
}

// ---- gfx1250 async global->LDS copy (ASYNCcnt) ----------------------------
#define AS1_ __attribute__((address_space(1)))
#define AS3_ __attribute__((address_space(3)))

__device__ __forceinline__ void async_cp16(const unsigned short* g,
                                           unsigned short* l) {
#if __has_builtin(__builtin_amdgcn_global_load_async_to_lds_b128)
  // signature: (v4i AS1* gsrc, v4i AS3* ldst, imm offset, imm cpol)
  AS1_ v4i_* gp = (AS1_ v4i_*)(unsigned long long)(size_t)g;
  AS3_ v4i_* lp = (AS3_ v4i_*)(unsigned)(size_t)l;  // low 32b == LDS offset
  __builtin_amdgcn_global_load_async_to_lds_b128(gp, lp, 0, 0);
#else
  const unsigned loff = (unsigned)(size_t)l;
  asm volatile("global_load_async_to_lds_b128 %0, %1, off"
               :: "v"(loff), "v"(g) : "memory");
#endif
}
__device__ __forceinline__ void wait_async0() {
#if __has_builtin(__builtin_amdgcn_s_wait_asynccnt)
  __builtin_amdgcn_s_wait_asynccnt(0);
#else
  asm volatile("s_wait_asynccnt 0" ::: "memory");
#endif
}

enum {
  F_RELU = 1, F_LN = 2, F_POS = 4, F_RES = 8,
  F_QKV = 16, F_WF = 32, F_WB = 64
};

// ---------------------------------------------------------------------------
// Weight transpose+convert:  Wt[l][n][k] = bf16(W[l][k][n])
// ---------------------------------------------------------------------------
__global__ __launch_bounds__(256) void wtranspose(
    const float* __restrict__ W, unsigned short* __restrict__ Wt, int K, int N) {
  const size_t base = (size_t)blockIdx.z * K * N;
  const int g = blockIdx.x * 256 + threadIdx.x;
  if (g >= K * N) return;
  const int n = g / K, k = g % K;
  Wt[base + (size_t)n * K + k] = f2bfu(W[base + (size_t)k * N + n]);
}

__global__ __launch_bounds__(256) void cvt_bf16(const float* __restrict__ in,
                                                unsigned short* __restrict__ o,
                                                int n) {
  const int g = blockIdx.x * 256 + threadIdx.x;
  if (g < n) o[g] = f2bfu(in[g]);
}

// ---------------------------------------------------------------------------
// Fused bf16-WMMA GEMM: 32 rows x 512 cols per 256-thread block.
// 8 waves x (2 row-frags x 4 col-tiles) accumulators = 8 WMMAs / K-step.
// A panel (32x32 bf16) double-buffered in LDS via async copies; B operand
// streamed from K-major bf16 weights with b128 loads.
// ---------------------------------------------------------------------------
template <int FLAGS>
__global__ __launch_bounds__(256) void gemm_wmma(
    const unsigned short* __restrict__ Xb, const unsigned short* __restrict__ Wt,
    const float* __restrict__ bias, const float* __restrict__ Res,
    const float* __restrict__ lns, const float* __restrict__ lnb,
    const float* __restrict__ pos, float* __restrict__ Yf,
    unsigned short* __restrict__ Yb, unsigned short* __restrict__ Vt,
    int K, int N) {
  __shared__ alignas(16) unsigned short aLDS[2][32][32];

  const int tid  = threadIdx.x;
  const int w    = tid >> 5;
  const int lane = tid & 31;
  const int r    = lane & 15;
  const int hi   = lane >> 4;
  const int row0 = blockIdx.y * 32;
  const int col0w = blockIdx.x * 512 + w * 64;

  v8f acc[2][4];
#pragma unroll
  for (int rt = 0; rt < 2; ++rt)
#pragma unroll
    for (int t = 0; t < 4; ++t)
      acc[rt][t] = (v8f){0.f,0.f,0.f,0.f,0.f,0.f,0.f,0.f};

  const unsigned short* wcol[4];
#pragma unroll
  for (int t = 0; t < 4; ++t) wcol[t] = Wt + (size_t)(col0w + t * 16 + r) * K;

  const int rr = tid >> 2, c4 = (tid & 3) * 8;  // async-copy lane mapping
  if (tid < 128)
    async_cp16(Xb + (size_t)(row0 + rr) * K + c4, &aLDS[0][rr][c4]);

  int buf = 0;
  for (int k0 = 0; k0 < K; k0 += 32) {
    wait_async0();        // issuers: panel[buf] landed; others: free
    __syncthreads();      // publish LDS panel to all waves
    if (tid < 128 && k0 + 32 < K)
      async_cp16(Xb + (size_t)(row0 + rr) * K + k0 + 32 + c4,
                 &aLDS[buf ^ 1][rr][c4]);
    if (k0 + 64 < K)
      __builtin_prefetch(wcol[0] + k0 + 64, 0, 1);  // global_prefetch_b8

    const v16bf a0 = make_frag(&aLDS[buf][r][0], hi);
    const v16bf a1 = make_frag(&aLDS[buf][16 + r][0], hi);
#pragma unroll
    for (int t = 0; t < 4; ++t) {
      const v16bf b = make_frag(wcol[t] + k0, hi);
      acc[0][t] = __builtin_amdgcn_wmma_f32_16x16x32_bf16(
          false, a0, false, b, (short)0, acc[0][t], false, false);
      acc[1][t] = __builtin_amdgcn_wmma_f32_16x16x32_bf16(
          false, a1, false, b, (short)0, acc[1][t], false, false);
    }
    buf ^= 1;
  }

  if constexpr (FLAGS & F_QKV) {
    // bf16 QK panel + V scattered transposed [b][h][dh][T]
#pragma unroll
    for (int rt = 0; rt < 2; ++rt)
#pragma unroll
      for (int t = 0; t < 4; ++t) {
        const int colg = col0w + t * 16 + r;
#pragma unroll
        for (int ii = 0; ii < 8; ++ii) {
          const int row = row0 + rt * 16 + ii + 8 * hi;
          const float v = acc[rt][t][ii] + bias[colg];
          if (colg < 2 * D_) {
            Yb[(size_t)row * N + colg] = f2bfu(v);
          } else {
            const int bb = row / T_, tt = row % T_;
            const int c = colg - 2 * D_, hh = c >> 6, dh = c & 63;
            Vt[(((size_t)bb * H_ + hh) * DH_ + dh) * T_ + tt] = f2bfu(v);
          }
        }
      }
  } else if constexpr (FLAGS & F_LN) {
    __shared__ float xr[32][512];
    __shared__ float psum[8][32], psq[8][32];
    __shared__ float mean_s[32], rstd_s[32];
#pragma unroll
    for (int rt = 0; rt < 2; ++rt)
#pragma unroll
      for (int t = 0; t < 4; ++t) {
        const int colg = col0w + t * 16 + r;
#pragma unroll
        for (int ii = 0; ii < 8; ++ii) {
          const int row = row0 + rt * 16 + ii + 8 * hi;
          float v = acc[rt][t][ii] + bias[colg];
          if constexpr (FLAGS & F_RES) v += Res[(size_t)row * N + colg];
          xr[rt * 16 + ii + 8 * hi][colg] = v;
        }
      }
    __syncthreads();
    const int lr = tid & 31, seg = tid >> 5;  // 32 rows x 8 col-segments
    float s = 0.f, q = 0.f;
#pragma unroll
    for (int c0 = 0; c0 < 64; ++c0) {
      const float v = xr[lr][seg * 64 + c0];
      s += v; q += v * v;
    }
    psum[seg][lr] = s; psq[seg][lr] = q;
    __syncthreads();
    if (tid < 32) {
      float su = 0.f, qu = 0.f;
#pragma unroll
      for (int k2 = 0; k2 < 8; ++k2) { su += psum[k2][tid]; qu += psq[k2][tid]; }
      const float mn  = su * (1.f / 512.f);
      const float var = qu * (1.f / 512.f) - mn * mn;
      mean_s[tid] = mn;
      rstd_s[tid] = rsqrtf(var + EPS_);
    }
    __syncthreads();
    const float mn = mean_s[lr], rs = rstd_s[lr];
    for (int c0 = 0; c0 < 64; ++c0) {
      const int c = seg * 64 + c0;
      const float val = (xr[lr][c] - mn) * rs * lns[c] + lnb[c];
      if constexpr (FLAGS & F_WF) Yf[(size_t)(row0 + lr) * N + c] = val;
      if constexpr (FLAGS & F_WB) Yb[(size_t)(row0 + lr) * N + c] = f2bfu(val);
    }
  } else {
#pragma unroll
    for (int rt = 0; rt < 2; ++rt)
#pragma unroll
      for (int t = 0; t < 4; ++t) {
        const int colg = col0w + t * 16 + r;
#pragma unroll
        for (int ii = 0; ii < 8; ++ii) {
          const int row = row0 + rt * 16 + ii + 8 * hi;
          float v = acc[rt][t][ii] + bias[colg];
          if constexpr (FLAGS & F_POS) v += pos[(size_t)(row % T_) * N + colg];
          if constexpr (FLAGS & F_RES) v += Res[(size_t)row * N + colg];
          if constexpr (FLAGS & F_RELU) v = fmaxf(v, 0.f);
          if constexpr (FLAGS & F_WF) Yf[(size_t)row * N + colg] = v;
          if constexpr (FLAGS & F_WB) Yb[(size_t)row * N + colg] = f2bfu(v);
        }
      }
  }
}

// ---------------------------------------------------------------------------
// Banded flash attention: one wave per (batch, head, 16-query tile).
// ---------------------------------------------------------------------------
__global__ __launch_bounds__(32) void attn_banded(
    const unsigned short* __restrict__ qkvb, const unsigned short* __restrict__ Vt,
    unsigned short* __restrict__ ctxb) {
  const int lane = threadIdx.x & 31;
  const int r = lane & 15, hi = lane >> 4;
  const int nqt = T_ / 16;
  const int bid = blockIdx.x;
  const int qt = bid % nqt;
  const int h  = (bid / nqt) % H_;
  const int b  = bid / (nqt * H_);
  const int i0 = qt * 16;
  const size_t rowbase = (size_t)b * T_;
  const int qoff = h * DH_, koff = D_ + h * DH_;
  const float scale = 0.125f;  // 1/sqrt(64)
  const unsigned bandw = PAST_ + FUT_;  // inclusive band width - 1

  __shared__ alignas(16) unsigned short pLDS[16][32];

  v16bf qa[2];
  {
    const unsigned short* qp = qkvb + (rowbase + i0 + r) * (size_t)(3 * D_) + qoff;
    qa[0] = make_frag(qp, hi);
    qa[1] = make_frag(qp + 32, hi);
  }
  const unsigned short* vp[4];
#pragma unroll
  for (int t = 0; t < 4; ++t)
    vp[t] = Vt + (((size_t)b * H_ + h) * DH_ + t * 16 + r) * T_;

  v8f O[4];
#pragma unroll
  for (int t = 0; t < 4; ++t) O[t] = (v8f){0.f,0.f,0.f,0.f,0.f,0.f,0.f,0.f};
  float mrow[8], lrow[8];
#pragma unroll
  for (int ii = 0; ii < 8; ++ii) { mrow[ii] = -1e30f; lrow[ii] = 0.f; }

  int jlo = (i0 - PAST_ < 0) ? 0 : (i0 - PAST_);
  jlo &= ~15;
  int jhi = i0 + 15 + FUT_;
  if (jhi > T_ - 1) jhi = T_ - 1;

  for (int j0 = jlo; j0 <= jhi; j0 += 32) {
    v8f s[2];
#pragma unroll
    for (int st = 0; st < 2; ++st) {
      s[st] = (v8f){0.f,0.f,0.f,0.f,0.f,0.f,0.f,0.f};
      int krow = j0 + st * 16 + r;
      if (krow > T_ - 1) krow = T_ - 1;   // clamped addr; masked below
      const unsigned short* kp = qkvb + (rowbase + krow) * (size_t)(3 * D_) + koff;
#pragma unroll
      for (int c = 0; c < 2; ++c) {
        const v16bf kb = make_frag(kp + c * 32, hi);
        s[st] = __builtin_amdgcn_wmma_f32_16x16x32_bf16(
            false, qa[c], false, kb, (short)0, s[st], false, false);
      }
    }
#pragma unroll
    for (int ii = 0; ii < 8; ++ii) {
      const int irow = i0 + ii + 8 * hi;
      const int ja = j0 + r, jb2 = j0 + 16 + r;
      // band: (unsigned)(j - (i-PAST)) <= PAST+FUT, plus j < T
      const bool ok0 = ((unsigned)(ja - (irow - PAST_)) <= bandw) & (ja < T_);
      const bool ok1 = ((unsigned)(jb2 - (irow - PAST_)) <= bandw) & (jb2 < T_);
      const float v0 = ok0 ? s[0][ii] * scale : -3.0e38f;
      const float v1 = ok1 ? s[1][ii] * scale : -3.0e38f;
      float mt = fmaxf(v0, v1);
      mt = fmaxf(mt, __shfl_xor(mt, 1));
      mt = fmaxf(mt, __shfl_xor(mt, 2));
      mt = fmaxf(mt, __shfl_xor(mt, 4));
      mt = fmaxf(mt, __shfl_xor(mt, 8));
      const float mnew = fmaxf(mrow[ii], mt);
      const float sc = __expf(mrow[ii] - mnew);
      mrow[ii] = mnew;
      const float p0 = __expf(v0 - mnew);
      const float p1 = __expf(v1 - mnew);
      float ssum = p0 + p1;
      ssum += __shfl_xor(ssum, 1);
      ssum += __shfl_xor(ssum, 2);
      ssum += __shfl_xor(ssum, 4);
      ssum += __shfl_xor(ssum, 8);
      lrow[ii] = lrow[ii] * sc + ssum;
#pragma unroll
      for (int t = 0; t < 4; ++t) O[t][ii] *= sc;
      pLDS[ii + 8 * hi][r]      = f2bfu(p0);
      pLDS[ii + 8 * hi][16 + r] = f2bfu(p1);
    }
    __syncthreads();
    const v16bf pa = make_frag(&pLDS[r][0], hi);
#pragma unroll
    for (int t = 0; t < 4; ++t) {
      const v16bf vb = make_frag(vp[t] + j0, hi);
      O[t] = __builtin_amdgcn_wmma_f32_16x16x32_bf16(
          false, pa, false, vb, (short)0, O[t], false, false);
    }
    __syncthreads();
  }
#pragma unroll
  for (int ii = 0; ii < 8; ++ii) {
    const float rinv = 1.0f / lrow[ii];
    const int row = i0 + ii + 8 * hi;
#pragma unroll
    for (int t = 0; t < 4; ++t)
      ctxb[(rowbase + row) * (size_t)D_ + h * DH_ + t * 16 + r] =
          f2bfu(O[t][ii] * rinv);
  }
}

// ---------------------------------------------------------------------------
__global__ __launch_bounds__(256) void meanpool(const float* __restrict__ emb,
                                                float* __restrict__ p) {
  const int g = blockIdx.x * 256 + threadIdx.x;  // 2048 threads total
  const int b = g >> 9, d = g & 511;
  const float* base = emb + (size_t)b * T_ * D_ + d;
  float s = 0.f;
  for (int t = 0; t < T_; ++t) s += base[(size_t)t * D_];
  p[g] = s * (1.0f / T_);
}

__global__ __launch_bounds__(256) void heads(
    const float* __restrict__ p, const float* __restrict__ mu1W,
    const float* __restrict__ mu1b, const float* __restrict__ mu2W,
    const float* __restrict__ mu2b, const float* __restrict__ lv1W,
    const float* __restrict__ lv1b, const float* __restrict__ lv2W,
    const float* __restrict__ lv2b, float* __restrict__ outMu,
    float* __restrict__ outLv) {
  __shared__ float ps[B_][D_];
  __shared__ float h1[2][B_][2 * LAT_];
  const int tid = threadIdx.x;
  for (int i = tid; i < B_ * D_; i += 256) ps[i >> 9][i & 511] = p[i];
  __syncthreads();
  for (int b = 0; b < B_; ++b) {
    float sm = mu1b[tid], sl = lv1b[tid];
    for (int k = 0; k < D_; ++k) {
      const float pv = ps[b][k];
      sm += pv * mu1W[k * (2 * LAT_) + tid];
      sl += pv * lv1W[k * (2 * LAT_) + tid];
    }
    h1[0][b][tid] = fmaxf(sm, 0.f);
    h1[1][b][tid] = fmaxf(sl, 0.f);
  }
  __syncthreads();
  if (tid < LAT_) {
    for (int b = 0; b < B_; ++b) {
      float sm = mu2b[tid], sl = lv2b[tid];
      for (int k = 0; k < 2 * LAT_; ++k) {
        sm += h1[0][b][k] * mu2W[k * LAT_ + tid];
        sl += h1[1][b][k] * lv2W[k * LAT_ + tid];
      }
      outMu[b * LAT_ + tid] = sm;
      outLv[b * LAT_ + tid] = sl;
    }
  }
}

// ---------------------------------------------------------------------------
extern "C" void kernel_launch(void* const* d_in, const int* in_sizes, int n_in,
                              void* d_out, int out_size, void* d_ws,
                              size_t ws_size, hipStream_t stream) {
  (void)in_sizes; (void)n_in; (void)out_size; (void)ws_size;
  const float* feat  = (const float*)d_in[0];
  const float* projW = (const float*)d_in[1];
  const float* projB = (const float*)d_in[2];
  const float* pos   = (const float*)d_in[3];
  const float* qkvW  = (const float*)d_in[4];
  const float* qkvB  = (const float*)d_in[5];
  const float* outW  = (const float*)d_in[6];
  const float* outB  = (const float*)d_in[7];
  const float* ln1s  = (const float*)d_in[8];
  const float* ln1b  = (const float*)d_in[9];
  const float* ff1W  = (const float*)d_in[10];
  const float* ff1b  = (const float*)d_in[11];
  const float* ff2W  = (const float*)d_in[12];
  const float* ff2b  = (const float*)d_in[13];
  const float* ln2s  = (const float*)d_in[14];
  const float* ln2b  = (const float*)d_in[15];
  const float* mu1W  = (const float*)d_in[16];
  const float* mu1b  = (const float*)d_in[17];
  const float* mu2W  = (const float*)d_in[18];
  const float* mu2b  = (const float*)d_in[19];
  const float* lv1W  = (const float*)d_in[20];
  const float* lv1b  = (const float*)d_in[21];
  const float* lv2W  = (const float*)d_in[22];
  const float* lv2b  = (const float*)d_in[23];

  float* emb   = (float*)d_out;                       // [8192 x 512] fp32
  float* outMu = emb + (size_t)M_ * D_;
  float* outLv = outMu + B_ * LAT_;

  float* x    = (float*)d_ws;                         // [8192 x 512] fp32
  float* pbuf = x + (size_t)M_ * D_;                  // [4 x 512]
  unsigned short* us = (unsigned short*)(pbuf + B_ * D_);
  unsigned short* xb     = us;  us += (size_t)M_ * D_;        // act, K-major
  unsigned short* qkvb   = us;  us += (size_t)M_ * 3 * D_;
  unsigned short* ctxb   = us;  us += (size_t)M_ * D_;
  unsigned short* hb     = us;  us += (size_t)M_ * FF_;
  unsigned short* vtb    = us;  us += (size_t)B_ * H_ * DH_ * T_ + 64;  // +pad
  unsigned short* projWt = us;  us += (size_t)D_ * IN_;
  unsigned short* qkvWt  = us;  us += (size_t)L_ * 3 * D_ * D_;
  unsigned short* outWt  = us;  us += (size_t)L_ * D_ * D_;
  unsigned short* ff1Wt  = us;  us += (size_t)L_ * FF_ * D_;
  unsigned short* ff2Wt  = us;  us += (size_t)L_ * D_ * FF_;

  const dim3 blk(256);
  // ---- one-time weight transpose/convert to bf16 [N][K] ----
  wtranspose<<<dim3((IN_ * D_ + 255) / 256, 1, 1),  blk, 0, stream>>>(projW, projWt, IN_, D_);
  wtranspose<<<dim3((D_ * 3 * D_ + 255) / 256, 1, L_), blk, 0, stream>>>(qkvW, qkvWt, D_, 3 * D_);
  wtranspose<<<dim3((D_ * D_ + 255) / 256, 1, L_),  blk, 0, stream>>>(outW, outWt, D_, D_);
  wtranspose<<<dim3((D_ * FF_ + 255) / 256, 1, L_), blk, 0, stream>>>(ff1W, ff1Wt, D_, FF_);
  wtranspose<<<dim3((FF_ * D_ + 255) / 256, 1, L_), blk, 0, stream>>>(ff2W, ff2Wt, FF_, D_);
  // bf16 copy of input features (K-major activation layout), parked in hb
  cvt_bf16<<<dim3((M_ * IN_ + 255) / 256), blk, 0, stream>>>(feat, hb, M_ * IN_);

  // x = feat @ projW + b + pos ; writes x (fp32) + xb (bf16)
  gemm_wmma<F_POS | F_WF | F_WB><<<dim3(1, M_ / 32), blk, 0, stream>>>(
      hb, projWt, projB, nullptr, nullptr, nullptr, pos, x, xb, nullptr, IN_, D_);

  for (int l = 0; l < L_; ++l) {
    gemm_wmma<F_QKV><<<dim3(3, M_ / 32), blk, 0, stream>>>(
        xb, qkvWt + (size_t)l * 3 * D_ * D_, qkvB + l * 3 * D_, nullptr,
        nullptr, nullptr, nullptr, nullptr, qkvb, vtb, D_, 3 * D_);
    attn_banded<<<dim3(B_ * H_ * (T_ / 16)), dim3(32), 0, stream>>>(qkvb, vtb, ctxb);
    gemm_wmma<F_LN | F_RES | F_WF | F_WB><<<dim3(1, M_ / 32), blk, 0, stream>>>(
        ctxb, outWt + (size_t)l * D_ * D_, outB + l * D_, x, ln1s + l * D_,
        ln1b + l * D_, nullptr, x, xb, nullptr, D_, D_);
    gemm_wmma<F_RELU | F_WB><<<dim3(4, M_ / 32), blk, 0, stream>>>(
        xb, ff1Wt + (size_t)l * FF_ * D_, ff1b + l * FF_, nullptr, nullptr,
        nullptr, nullptr, nullptr, hb, nullptr, D_, FF_);
    float* Yout = (l == L_ - 1) ? emb : x;
    gemm_wmma<F_LN | F_RES | F_WF | F_WB><<<dim3(1, M_ / 32), blk, 0, stream>>>(
        hb, ff2Wt + (size_t)l * D_ * FF_, ff2b + l * D_, x, ln2s + l * D_,
        ln2b + l * D_, nullptr, Yout, xb, nullptr, FF_, D_);
  }

  meanpool<<<dim3((B_ * D_) / 256), blk, 0, stream>>>(emb, pbuf);
  heads<<<dim3(1), blk, 0, stream>>>(pbuf, mu1W, mu1b, mu2W, mu2b, lv1W, lv1b,
                                     lv2W, lv2b, outMu, outLv);
}
